// UGCA_18940805775522
// MI455X (gfx1250) — matrix operand order repaired and checked by reading, not compiled, not run on hardware
//
#include <hip/hip_runtime.h>
#include <stdint.h>

// ---------------------------------------------------------------------------
// CDNA5 WMMA (wave32): D(16x16 f32) = A(16x32 f16) x B(32x16 f16) + C
// LDS tiles are stored pre-swizzled in fragment order: each lane's 16 halves
// are contiguous (32B aligned) so a fragment load is one v16h LDS read.
// Attention V chunks are double-buffered through LDS with CDNA5 async
// global->LDS loads (ASYNCcnt), overlapping HBM latency with WMMA issue.
// ---------------------------------------------------------------------------
typedef __attribute__((ext_vector_type(16))) _Float16 v16h;
typedef __attribute__((ext_vector_type(4)))  _Float16 v4h;
typedef __attribute__((ext_vector_type(8)))  float    v8f;

#define NEG_BIG (-1e30f)

__device__ inline float softplusf(float x) {
  return x > 20.0f ? x : log1pf(expf(x));
}

// A-fragment element position (ISA 7.12.2, 16-bit A 16x32):
//   lane = m | ((k&8)<<1) ; elem = (k&7) | ((k&16)>>1)
__device__ inline int a_sw_idx(int m, int k) {
  return ((m & 15) | ((k & 8) << 1)) * 16 + ((k & 7) | ((k & 16) >> 1));
}
// B-fragment element position (32x16): lane = n | ((k&8)<<1) ; elem = (k&7) | ((k&16)>>1)
__device__ inline int b_sw_idx(int k, int n) {
  return ((n & 15) | ((k & 8) << 1)) * 16 + ((k & 7) | ((k & 16) >> 1));
}
__device__ inline v16h frag_ld(const _Float16* s, int lane) {
  return *(const v16h*)(s + lane * 16);
}
__device__ inline v4h cvt4(float x, float y, float z, float w) {
  v4h p; p[0] = (_Float16)x; p[1] = (_Float16)y; p[2] = (_Float16)z; p[3] = (_Float16)w;
  return p;
}

// CDNA5 async global->LDS: per-lane 16B, tracked by ASYNCcnt (CDNA5 ISA).
// dst_lds: per-lane LDS byte address (VDST); goff: 32-bit byte offset added to
// the 64-bit SGPR base (GVS mode).
__device__ inline void async_load_b128(uint32_t dst_lds, uint32_t goff,
                                       const void* sbase) {
  asm volatile("global_load_async_to_lds_b128 %0, %1, %2"
               :: "v"(dst_lds), "v"(goff), "s"(sbase) : "memory");
}

// ---------------------------------------------------------------------------
// 1) NIG head -> per-token gate (pre-topk), EDL partial sums per block
// ---------------------------------------------------------------------------
__global__ __launch_bounds__(256)
void nig_gate(const float* __restrict__ h, const unsigned char* __restrict__ mask,
              const float* __restrict__ w, const float* __restrict__ bvec,
              float* __restrict__ g_raw, float* __restrict__ edl_part, int Ntok) {
  __shared__ float red[256];
  const int t = blockIdx.x * 256 + threadIdx.x;
  float ab = 0.0f;
  if (t < Ntok) {
    const float* hp = h + (size_t)t * 256;
    float r1 = bvec[1], r2 = bvec[2], r3 = bvec[3];
    for (int d4 = 0; d4 < 64; ++d4) {
      const float4 h4 = *(const float4*)(hp + d4 * 4);
      const float4 w0 = *(const float4*)(w + (d4 * 4 + 0) * 4);
      const float4 w1 = *(const float4*)(w + (d4 * 4 + 1) * 4);
      const float4 w2 = *(const float4*)(w + (d4 * 4 + 2) * 4);
      const float4 w3 = *(const float4*)(w + (d4 * 4 + 3) * 4);
      r1 += h4.x * w0.y + h4.y * w1.y + h4.z * w2.y + h4.w * w3.y;
      r2 += h4.x * w0.z + h4.y * w1.z + h4.z * w2.z + h4.w * w3.z;
      r3 += h4.x * w0.w + h4.y * w1.w + h4.z * w2.w + h4.w * w3.w;
    }
    float nu    = softplusf(r1);
    float spa   = softplusf(r2);       // alpha = 1 + spa
    float beta  = softplusf(r3);
    float s2    = beta / (nu * spa + 1e-6f);
    float g     = expf(-4.0f * s2);
    g = fminf(fmaxf(g, 1e-4f), 1.0f);
    g *= mask[t] ? 1.0f : 0.0f;
    g_raw[t] = g;
    ab = (1.0f + spa) + beta;          // alpha + beta
  }
  red[threadIdx.x] = ab;
  __syncthreads();
  for (int s = 128; s; s >>= 1) {
    if (threadIdx.x < s) red[threadIdx.x] += red[threadIdx.x + s];
    __syncthreads();
  }
  if (threadIdx.x == 0) edl_part[blockIdx.x] = red[0];
}

// ---------------------------------------------------------------------------
// 2) Per-row variable top-k (rank by pairwise count, stable on ties) + log g
// ---------------------------------------------------------------------------
__global__ __launch_bounds__(256)
void topk_gate(const float* __restrict__ g_raw, const unsigned char* __restrict__ mask,
               float* __restrict__ g_out, float* __restrict__ logg,
               float* __restrict__ gsum_row, float* __restrict__ msum_row, int N) {
  __shared__ float sg[1024];
  __shared__ float red[256];
  const int row = blockIdx.x, tid = threadIdx.x;
  float msum = 0.0f;
  for (int i = tid; i < N; i += 256) {
    sg[i] = g_raw[row * N + i];
    msum += mask[row * N + i] ? 1.0f : 0.0f;
  }
  red[tid] = msum;
  __syncthreads();
  for (int s = 128; s; s >>= 1) {
    if (tid < s) red[tid] += red[tid + s];
    __syncthreads();
  }
  const float mtot = red[0];
  const int k = (int)fmaxf(mtot * 0.5f, 1.0f);   // floor via truncation
  __syncthreads();

  float gsum = 0.0f;
  for (int i = tid; i < N; i += 256) {
    const float gi = sg[i];
    int rank = 0;
    for (int j = 0; j < N; ++j) {
      float gj = sg[j];
      rank += (gj > gi || (gj == gi && j < i)) ? 1 : 0;
    }
    float go = (rank < k) ? gi : 0.0f;
    g_out[row * N + i] = go;
    logg[row * N + i]  = logf(go + 1e-8f);
    gsum += go;
  }
  __syncthreads();
  red[tid] = gsum;
  __syncthreads();
  for (int s = 128; s; s >>= 1) {
    if (tid < s) red[tid] += red[tid + s];
    __syncthreads();
  }
  if (tid == 0) { gsum_row[row] = red[0]; msum_row[row] = mtot; }
}

// ---------------------------------------------------------------------------
// 3) Regularizer loss (serial, deterministic)
// ---------------------------------------------------------------------------
__global__ void loss_kernel(const float* edlD, const float* edlP,
                            const float* gsD, const float* msD,
                            const float* gsP, const float* msP,
                            float* __restrict__ out) {
  float eD = 0, eP = 0, gD = 0, mD = 0, gP = 0, mP = 0;
  for (int i = 0; i < 16; ++i) eD += edlD[i];
  for (int i = 0; i < 32; ++i) eP += edlP[i];
  for (int i = 0; i < 8; ++i) { gD += gsD[i]; mD += msD[i]; gP += gsP[i]; mP += msP[i]; }
  float dD = gD / (mD + 1e-8f) - 0.3f;
  float dP = gP / (mP + 1e-8f) - 0.3f;
  out[0] = 0.01f * dD * dD + 0.01f * dP * dP
         + 1e-3f * (eD / 4096.0f) + 1e-3f * (eP / 8192.0f);
}

// ---------------------------------------------------------------------------
// 4) WMMA GEMM: C[M,N] = A[M,K] * W[K,N] + bias (+ residual if HASR)
//    64x32 block tile, 8 waves x one 16x16 WMMA tile, K-step 32.
//    LDS tiles stored pre-swizzled; global loads are float4 (b128);
//    next K-step tiles prefetched (global_prefetch_b8).
// ---------------------------------------------------------------------------
template <bool HASR>
__global__ __launch_bounds__(256)
void gemm16(const float* __restrict__ A, const float* __restrict__ W,
            const float* __restrict__ bias, const float* __restrict__ R,
            float* __restrict__ C, int M, int N, int K) {
  __shared__ _Float16 As_sw[4][512];   // 4 swizzled A tiles (16x32)
  __shared__ _Float16 Bs_sw[2][512];   // 2 swizzled B tiles (32x16)
  const int m0   = blockIdx.x * 64;
  const int n0   = blockIdx.y * 32;
  const int tid  = threadIdx.x;
  const int lane = tid & 31;
  const int wave = tid >> 5;         // 0..7
  const int wm   = wave & 3;         // m-tile 0..3
  const int wn   = wave >> 2;        // n-tile 0..1
  v8f acc = {};
  for (int k0 = 0; k0 < K; k0 += 32) {
    // A tile: 64 rows x 32 k = 512 float4 groups, 2 per thread.
#pragma unroll
    for (int t = 0; t < 2; ++t) {
      int i = tid + t * 256;             // 0..511
      int r = i >> 3, c4 = i & 7;        // row, k-group
      const float4 a4 = *(const float4*)(A + (size_t)(m0 + r) * K + k0 + c4 * 4);
      int kb = c4 * 4;
      int li = (r & 15) | ((kb & 8) << 1);
      int e0 = (kb & 7) | ((kb & 16) >> 1);
      *(v4h*)(&As_sw[r >> 4][li * 16 + e0]) = cvt4(a4.x, a4.y, a4.z, a4.w);
    }
    // B tile: 32 k x 32 n = 256 float4 groups, 1 per thread.
    {
      int r = tid >> 3, c4 = tid & 7;    // k row, n-group
      const float4 w4 = *(const float4*)(W + (size_t)(k0 + r) * N + n0 + c4 * 4);
      int nb = c4 * 4, nt = nb >> 4;
      Bs_sw[nt][b_sw_idx(r, nb + 0)] = (_Float16)w4.x;
      Bs_sw[nt][b_sw_idx(r, nb + 1)] = (_Float16)w4.y;
      Bs_sw[nt][b_sw_idx(r, nb + 2)] = (_Float16)w4.z;
      Bs_sw[nt][b_sw_idx(r, nb + 3)] = (_Float16)w4.w;
    }
    // Prefetch next K-step tiles while this step is consumed.
    if (k0 + 32 < K) {
      __builtin_prefetch(A + (size_t)(m0 + (tid >> 2)) * K + k0 + 32, 0, 3);
      __builtin_prefetch(W + (size_t)(k0 + 32 + (tid & 31)) * N + n0, 0, 3);
    }
    __syncthreads();
    v16h a = frag_ld(As_sw[wm], lane);
    v16h b = frag_ld(Bs_sw[wn], lane);
    acc = __builtin_amdgcn_wmma_f32_16x16x32_f16(false, a, false, b,
                                                 (short)0, acc, false, false);
    __syncthreads();
  }
  const int col = lane & 15;
  const int rb  = (lane & 16) ? 8 : 0;
  const int gn  = n0 + wn * 16 + col;
  const float bv = bias[gn];
#pragma unroll
  for (int r = 0; r < 8; ++r) {
    int gm = m0 + wm * 16 + rb + r;
    float v = acc[r] + bv;
    if (HASR) v += R[(size_t)gm * N + gn];
    C[(size_t)gm * N + gn] = v;
  }
}

// ---------------------------------------------------------------------------
// 5) Attention: one wave per (b, h, 16-query tile). DH=32 == WMMA K:
//    QK^T = one WMMA / 16 keys; ctx = P*V = 2 WMMAs / 32 keys.
//    Q fragment is loaded to registers once; scores live in swizzled-A chunk
//    layout; V chunks are double-buffered via async global->LDS (ASYNCcnt).
// ---------------------------------------------------------------------------
__global__ __launch_bounds__(32)
void attn16(const float* __restrict__ Q, const float* __restrict__ Kb,
            const float* __restrict__ Vb, const unsigned char* __restrict__ mask,
            const float* __restrict__ logg, float* __restrict__ Ctx,
            int Nq, int Nk) {
  extern __shared__ _Float16 smem_h[];
  _Float16* sc = smem_h;          // 16*Nk probs, (Nk/32) chunks of 512 (swizzled A)
  _Float16* qs = sc + 16 * Nk;    // 512: swizzled A tile of Q
  _Float16* kv = qs + 512;        // 1024: swizzled B tiles for K^T / V chunks
  float*    vraw = (float*)(kv + 1024);  // 2 x 1024 f32: async V double buffer

  const int lane = threadIdx.x;
  const int nqt  = Nq / 16;
  const int qt   = blockIdx.x % nqt;
  const int h    = (blockIdx.x / nqt) & 7;
  const int b    = blockIdx.x / (nqt * 8);
  const int q0   = qt * 16;
  const float scale = 0.17677669529663687f;   // 1/sqrt(32)

  // ---- Q tile -> swizzled LDS -> registers (once) --------------------------
#pragma unroll
  for (int t = 0; t < 4; ++t) {
    int i = lane + t * 32;               // 0..127
    int m = i >> 3, d4 = i & 7;
    const float4 q4 = *(const float4*)(Q + ((size_t)(b * Nq + q0 + m)) * 256 + h * 32 + d4 * 4);
    int kb = d4 * 4;
    int li = m | ((kb & 8) << 1);
    int e0 = (kb & 7) | ((kb & 16) >> 1);
    *(v4h*)(&qs[li * 16 + e0]) = cvt4(q4.x, q4.y, q4.z, q4.w);
  }
  __syncthreads();
  const v16h aq = frag_ld(qs, lane);

  // ---- scores = QK^T * scale, masked_fill, + log_g -------------------------
  for (int kt = 0; kt < Nk; kt += 16) {
    if (kt + 16 < Nk)
      __builtin_prefetch(Kb + ((size_t)(b * Nk + kt + 16 + (lane >> 1))) * 256 + h * 32, 0, 3);
#pragma unroll
    for (int t = 0; t < 4; ++t) {
      int i = lane + t * 32;             // 0..127
      int key = i >> 3, d4 = i & 7;      // B[k=dh][n=key]
      const float4 k4 = *(const float4*)(Kb + ((size_t)(b * Nk + kt + key)) * 256 + h * 32 + d4 * 4);
      int kb = d4 * 4;
      int li = key | ((kb & 8) << 1);
      int e0 = (kb & 7) | ((kb & 16) >> 1);
      *(v4h*)(&kv[li * 16 + e0]) = cvt4(k4.x, k4.y, k4.z, k4.w);
    }
    __syncthreads();
    v8f s = {};
    s = __builtin_amdgcn_wmma_f32_16x16x32_f16(false, aq, false, frag_ld(kv, lane),
                                               (short)0, s, false, false);
    const int col = lane & 15;
    const int key = kt + col;
    const float add = mask[b * Nk + key] ? 0.0f : NEG_BIG;
    const float lg  = logg[b * Nk + key];
    const int rb = (lane & 16) ? 8 : 0;
    _Float16* schunk = sc + (key >> 5) * 512;
    const int klocal = key & 31;
#pragma unroll
    for (int r = 0; r < 8; ++r)
      schunk[a_sw_idx(rb + r, klocal)] = (_Float16)(s[r] * scale + add + lg);
    __syncthreads();
  }

  // ---- row softmax (lanes 0..15, one row each; swizzled addressing) --------
  if (lane < 16) {
    float mx = -3.4e38f;
    for (int j = 0; j < Nk; ++j)
      mx = fmaxf(mx, (float)sc[(j >> 5) * 512 + a_sw_idx(lane, j & 31)]);
    float sum = 0.0f;
    for (int j = 0; j < Nk; ++j) {
      int idx = (j >> 5) * 512 + a_sw_idx(lane, j & 31);
      float e = expf((float)sc[idx] - mx);
      sc[idx] = (_Float16)e;
      sum += e;
    }
    float inv = 1.0f / sum;
    for (int j = 0; j < Nk; ++j) {
      int idx = (j >> 5) * 512 + a_sw_idx(lane, j & 31);
      sc[idx] = (_Float16)((float)sc[idx] * inv);
    }
  }
  __syncthreads();

  // ---- ctx = P * V, V chunks streamed with async global->LDS ---------------
  const uint32_t vraw_lds = (uint32_t)(uintptr_t)vraw;   // LDS byte address
  const float*   vbase    = Vb + (size_t)b * Nk * 256 + h * 32;

  // issue one 32-key x 32-dh raw f32 chunk (8 x b128 per lane) into buffer
#define ISSUE_V_CHUNK(kc_, buf_)                                             \
  {                                                                          \
    uint32_t dst0 = vraw_lds + (uint32_t)(buf_) * 4096u;                     \
    _Pragma("unroll")                                                        \
    for (int t = 0; t < 8; ++t) {                                            \
      int g = lane + t * 32;                                                 \
      int key = g >> 3, d4 = g & 7;                                          \
      uint32_t goff = (uint32_t)(((kc_) + key) * 256 + d4 * 4) * 4u;         \
      async_load_b128(dst0 + (uint32_t)g * 16u, goff, vbase);                \
    }                                                                        \
  }

  ISSUE_V_CHUNK(0, 0);
  v8f c0 = {}, c1 = {};
  int buf = 0;
  for (int kc = 0; kc < Nk; kc += 32, buf ^= 1) {
    const bool has_next = (kc + 32 < Nk);
    if (has_next) {
      ISSUE_V_CHUNK(kc + 32, buf ^ 1);
      asm volatile("s_wait_asynccnt 8" ::: "memory");   // current chunk done
    } else {
      asm volatile("s_wait_asynccnt 0" ::: "memory");   // drain
    }
    __syncthreads();
    // convert raw f32 chunk -> swizzled f16 B tiles (two dh halves)
    const float* src = vraw + buf * 1024;
#pragma unroll
    for (int t = 0; t < 8; ++t) {
      int g = lane + t * 32;
      int key = g >> 3, d4 = g & 7;
      const float4 v4 = *(const float4*)(src + g * 4);
      int dh = d4 * 4, nt = dh >> 4;
      kv[nt * 512 + b_sw_idx(key, dh + 0)] = (_Float16)v4.x;
      kv[nt * 512 + b_sw_idx(key, dh + 1)] = (_Float16)v4.y;
      kv[nt * 512 + b_sw_idx(key, dh + 2)] = (_Float16)v4.z;
      kv[nt * 512 + b_sw_idx(key, dh + 3)] = (_Float16)v4.w;
    }
    __syncthreads();
    v16h ap = frag_ld(sc + kc * 16, lane);   // chunk (kc/32)*512
    c0 = __builtin_amdgcn_wmma_f32_16x16x32_f16(false, ap, false, frag_ld(kv, lane),
                                                (short)0, c0, false, false);
    c1 = __builtin_amdgcn_wmma_f32_16x16x32_f16(false, ap, false, frag_ld(kv + 512, lane),
                                                (short)0, c1, false, false);
    __syncthreads();
  }
#undef ISSUE_V_CHUNK

  const int col = lane & 15;
  const int rb  = (lane & 16) ? 8 : 0;
#pragma unroll
  for (int r = 0; r < 8; ++r) {
    size_t base = ((size_t)(b * Nq + q0 + rb + r)) * 256 + h * 32;
    Ctx[base + col]      = c0[r];
    Ctx[base + 16 + col] = c1[r];
  }
}

// ---------------------------------------------------------------------------
// 6) LayerNorm, one wave per token, D=256, float4 I/O
// ---------------------------------------------------------------------------
__global__ __launch_bounds__(32)
void lnorm(const float* __restrict__ X, const float* __restrict__ g,
           const float* __restrict__ bvec, float* __restrict__ Y, int M) {
  const int t = blockIdx.x;
  const int lane = threadIdx.x;
  const float* x = X + (size_t)t * 256;
  const float4 x0 = *(const float4*)(x + lane * 4);
  const float4 x1 = *(const float4*)(x + 128 + lane * 4);
  float vals[8] = {x0.x, x0.y, x0.z, x0.w, x1.x, x1.y, x1.z, x1.w};
  float s = 0.0f;
#pragma unroll
  for (int i = 0; i < 8; ++i) s += vals[i];
  for (int o = 16; o; o >>= 1) s += __shfl_xor(s, o, 32);
  const float mean = s * (1.0f / 256.0f);
  float v = 0.0f;
#pragma unroll
  for (int i = 0; i < 8; ++i) { float d = vals[i] - mean; v += d * d; }
  for (int o = 16; o; o >>= 1) v += __shfl_xor(v, o, 32);
  const float rstd = rsqrtf(v * (1.0f / 256.0f) + 1e-5f);
  const float4 g0 = *(const float4*)(g + lane * 4);
  const float4 g1 = *(const float4*)(g + 128 + lane * 4);
  const float4 b0 = *(const float4*)(bvec + lane * 4);
  const float4 b1 = *(const float4*)(bvec + 128 + lane * 4);
  float4 y0, y1;
  y0.x = (vals[0] - mean) * rstd * g0.x + b0.x;
  y0.y = (vals[1] - mean) * rstd * g0.y + b0.y;
  y0.z = (vals[2] - mean) * rstd * g0.z + b0.z;
  y0.w = (vals[3] - mean) * rstd * g0.w + b0.w;
  y1.x = (vals[4] - mean) * rstd * g1.x + b1.x;
  y1.y = (vals[5] - mean) * rstd * g1.y + b1.y;
  y1.z = (vals[6] - mean) * rstd * g1.z + b1.z;
  y1.w = (vals[7] - mean) * rstd * g1.w + b1.w;
  float* y = Y + (size_t)t * 256;
  *(float4*)(y + lane * 4) = y0;
  *(float4*)(y + 128 + lane * 4) = y1;
}

// ---------------------------------------------------------------------------
// Host-side launch
// ---------------------------------------------------------------------------
static void run_mhca(const float* qx, const float* kx,
                     const unsigned char* mask_k, const float* log_g,
                     int Nq, int Nk,
                     const float* Wq, const float* bq, const float* Wk, const float* bk,
                     const float* Wv, const float* bv, const float* Wo, const float* bo,
                     const float* ln_g, const float* ln_b, float* out_x,
                     float* qb, float* kb, float* vb, float* cb, hipStream_t stream) {
  const int Mq = 8 * Nq, Mk = 8 * Nk;
  gemm16<false><<<dim3(Mq / 64, 8), 256, 0, stream>>>(qx, Wq, bq, nullptr, qb, Mq, 256, 256);
  gemm16<false><<<dim3(Mk / 64, 8), 256, 0, stream>>>(kx, Wk, bk, nullptr, kb, Mk, 256, 256);
  gemm16<false><<<dim3(Mk / 64, 8), 256, 0, stream>>>(kx, Wv, bv, nullptr, vb, Mk, 256, 256);
  // sc(16*Nk f16) + qs(512 f16) + kv(1024 f16) + vraw(2*1024 f32)
  size_t shm = (size_t)16 * Nk * sizeof(_Float16) + 512 * 2 + 1024 * 2 + 2 * 4096;
  attn16<<<8 * 8 * (Nq / 16), 32, shm, stream>>>(qb, kb, vb, mask_k, log_g, cb, Nq, Nk);
  // y = ctx @ Wo + bo + residual(qx)  (qb reused as y)
  gemm16<true><<<dim3(Mq / 64, 8), 256, 0, stream>>>(cb, Wo, bo, qx, qb, Mq, 256, 256);
  lnorm<<<Mq, 32, 0, stream>>>(qb, ln_g, ln_b, out_x, Mq);
}

extern "C" void kernel_launch(void* const* d_in, const int* in_sizes, int n_in,
                              void* d_out, int out_size, void* d_ws, size_t ws_size,
                              hipStream_t stream) {
  (void)in_sizes; (void)n_in; (void)out_size; (void)ws_size;
  const float* H_D  = (const float*)d_in[0];
  const unsigned char* mask_D = (const unsigned char*)d_in[1];
  const float* H_P  = (const float*)d_in[2];
  const unsigned char* mask_P = (const unsigned char*)d_in[3];
  const float* nig_w = (const float*)d_in[4];
  const float* nig_b = (const float*)d_in[5];
  const float* Wq = (const float*)d_in[6];  const float* bq = (const float*)d_in[7];
  const float* Wk = (const float*)d_in[8];  const float* bk = (const float*)d_in[9];
  const float* Wv = (const float*)d_in[10]; const float* bv = (const float*)d_in[11];
  const float* Wo = (const float*)d_in[12]; const float* bo = (const float*)d_in[13];
  const float* ln_g = (const float*)d_in[14]; const float* ln_b = (const float*)d_in[15];

  float* out  = (float*)d_out;
  float* Xd   = out;                  // [8,512,256]
  float* Xp   = out + 1048576;        // [8,1024,256]
  float* gD   = out + 3145728;        // [8,512]
  float* gP   = out + 3149824;        // [8,1024]
  float* loss = out + 3158016;        // scalar

  float* ws       = (float*)d_ws;
  float* g_raw_D  = ws;               // 4096
  float* g_raw_P  = ws + 4096;        // 8192
  float* log_g_D  = ws + 12288;       // 4096
  float* log_g_P  = ws + 16384;       // 8192
  float* edl_D    = ws + 24576;       // 16 block partials
  float* edl_P    = ws + 24592;       // 32 block partials
  float* gsum_D   = ws + 24624;       // 8
  float* msum_D   = ws + 24632;       // 8
  float* gsum_P   = ws + 24640;       // 8
  float* msum_P   = ws + 24648;       // 8
  float* qb       = ws + 24704;       // 4 x 8192*256 floats (~32MB)
  float* kb       = qb + 2097152;
  float* vb       = kb + 2097152;
  float* cb       = vb + 2097152;

  // Gating path
  nig_gate<<<16, 256, 0, stream>>>(H_D, mask_D, nig_w, nig_b, g_raw_D, edl_D, 4096);
  nig_gate<<<32, 256, 0, stream>>>(H_P, mask_P, nig_w, nig_b, g_raw_P, edl_P, 8192);
  topk_gate<<<8, 256, 0, stream>>>(g_raw_D, mask_D, gD, log_g_D, gsum_D, msum_D, 512);
  topk_gate<<<8, 256, 0, stream>>>(g_raw_P, mask_P, gP, log_g_P, gsum_P, msum_P, 1024);
  loss_kernel<<<1, 1, 0, stream>>>(edl_D, edl_P, gsum_D, msum_D, gsum_P, msum_P, loss);

  // Xd = MHCA(H_D <- H_P), then Xp = MHCA(H_P <- Xd)
  run_mhca(H_D, H_P, mask_P, log_g_P, 512, 1024,
           Wq, bq, Wk, bk, Wv, bv, Wo, bo, ln_g, ln_b, Xd, qb, kb, vb, cb, stream);
  run_mhca(H_P, Xd, mask_D, log_g_D, 1024, 512,
           Wq, bq, Wk, bk, Wv, bv, Wo, bo, ln_g, ln_b, Xp, qb, kb, vb, cb, stream);
}
